// Network_66898410602546
// MI455X (gfx1250) — matrix operand order repaired and checked by reading, not compiled, not run on hardware
//
#include <hip/hip_runtime.h>
#include <hip/hip_bf16.h>

typedef __attribute__((ext_vector_type(16))) _Float16 v16h;
typedef __attribute__((ext_vector_type(8)))  _Float16 v8h;
typedef __attribute__((ext_vector_type(8)))  float    v8f;
typedef __attribute__((ext_vector_type(4)))  float    v4f;

#define NB 512
#define NA 32
#define NE 64
#define NK 32
#define NL0 64
#define NL1 16
#define NSG 80
#define NOUT 112   // L0 + 3*L1

// ---- LDS layout (bytes), all 16B aligned ----
constexpr int OFF_G    = 0;        // f16 [a][n][k]        32*32*32*2 = 65536
constexpr int OFF_RHAT = 65536;    // f32 [a][x][n]        32*3*32*4  = 12288
constexpr int OFF_S    = 77824;    // f32 sT[c][n]         64*32*4    = 8192
constexpr int OFF_V    = 86016;    // f32 vT[x][c16][n]    3*16*32*4  = 6144
constexpr int OFF_WCAT = 92160;    // f16 WcatT[c][k]      160*32*2   = 10240
constexpr int OFF_PSS  = 102400;   // f16 Pss[a][c64]      32*64*2    = 4096
constexpr int OFF_PHV  = 106496;   // f16 Phv[x][a][c64]   3*32*64*2  = 12288
constexpr int OFF_PVV  = 118784;   // f16 Pvv[x][a][c32p]  3*32*32*2  = 6144 (k-padded)
constexpr int OFF_PVS  = 124928;   // f16 Pvs[a][c32p]     32*32*2    = 2048 (k-padded)
constexpr int OFF_WMSS = 126976;   // f16 WmixT_ss[co][ci] 80*64*2    = 10240
constexpr int OFF_WMSV = 137216;   // f16 WmixT_sv[co][ci] 16*64*2    = 2048
constexpr int OFF_WMVV = 139264;   // f16 WmixT_vv[co][32] 16*32*2    = 1024 (padded)
constexpr int OFF_WMVS = 140288;   // f16 WmixT_vs[co][32] 80*32*2    = 5120 (padded)
constexpr int OFF_SOUT = 145408;   // f32 s_out[a][80]     32*80*4    = 10240
constexpr int OFF_VOUT = 155648;   // f32 v_out[x][a][c16] 3*32*16*4  = 6144
constexpr int OFF_GEO  = 161792;   // f32 [32*3]                     = 384
constexpr int OFF_MASK = 162176;   // f32 [32]                       = 128
constexpr int SMEM_BYTES = 162304;

__device__ __forceinline__ float sigmoidf_(float x) { return 1.0f / (1.0f + __expf(-x)); }
__device__ __forceinline__ float siluf_(float x)    { return x * sigmoidf_(x); }

union F8 { v4f q[2]; float f[8]; };

__device__ __forceinline__ F8 ld_f8(const float* p) {
  F8 r;
  r.q[0] = *(const v4f*)p;
  r.q[1] = *(const v4f*)(p + 4);
  return r;
}

__device__ __forceinline__ v16h ld_ab(const _Float16* p0, const _Float16* p1) {
  union { v16h v; v8h h[2]; } u;
  u.h[0] = *(const v8h*)p0;
  u.h[1] = *(const v8h*)p1;
  return u.v;
}

__device__ __forceinline__ v8f wmma_f16(v16h a, v16h b, v8f c) {
  // D = A(16x32 f16) * B(32x16 f16) + C(16x16 f32)
  return __builtin_amdgcn_wmma_f32_16x16x32_f16(false, a, false, b, (short)0, c, false, false);
}

__global__ __launch_bounds__(256)
void painn2_kernel(const int* __restrict__ Z, const float* __restrict__ geo,
                   const float* __restrict__ maskg, const float* __restrict__ emb,
                   const float* __restrict__ Wr_ss1, const float* __restrict__ Wmix_ss1,
                   const float* __restrict__ Wr_sv1, const float* __restrict__ Wmix_sv1,
                   const float* __restrict__ Wr_ss2, const float* __restrict__ Wmix_ss2,
                   const float* __restrict__ Wr_sv2, const float* __restrict__ Wmix_sv2,
                   const float* __restrict__ Wr_vv2, const float* __restrict__ Wmix_vv2,
                   const float* __restrict__ Wr_vs2, const float* __restrict__ Wmix_vs2,
                   float* __restrict__ out)
{
  extern __shared__ char smem[];
  _Float16* hG    = (_Float16*)(smem + OFF_G);
  float*    hRhat = (float*)(smem + OFF_RHAT);
  float*    hSf   = (float*)(smem + OFF_S);
  float*    hVf   = (float*)(smem + OFF_V);
  _Float16* hWcat = (_Float16*)(smem + OFF_WCAT);
  _Float16* hPss  = (_Float16*)(smem + OFF_PSS);
  _Float16* hPhv  = (_Float16*)(smem + OFF_PHV);
  _Float16* hPvv  = (_Float16*)(smem + OFF_PVV);
  _Float16* hPvs  = (_Float16*)(smem + OFF_PVS);
  _Float16* hWmSS = (_Float16*)(smem + OFF_WMSS);
  _Float16* hWmSV = (_Float16*)(smem + OFF_WMSV);
  _Float16* hWmVV = (_Float16*)(smem + OFF_WMVV);
  _Float16* hWmVS = (_Float16*)(smem + OFF_WMVS);
  float*    hSoutF = (float*)(smem + OFF_SOUT);
  float*    hVoutF = (float*)(smem + OFF_VOUT);
  float*    hGeo   = (float*)(smem + OFF_GEO);
  float*    hMask  = (float*)(smem + OFF_MASK);

  const int b    = blockIdx.x;
  const int t    = threadIdx.x;
  const int lane = t & 31;
  const int w    = t >> 5;          // wave id (8 waves)
  const int m    = lane & 15;       // tile row/col within lane group
  const int h    = lane >> 4;       // lane half (wave32)
  const int kh   = h * 8;           // A-operand K-half offset

  const v8f z8 = {0.f, 0.f, 0.f, 0.f, 0.f, 0.f, 0.f, 0.f};

  // ======================= Phase A: geometry, RBF, staging =======================
  for (int i = t; i < NA * 3; i += 256) hGeo[i] = geo[b * NA * 3 + i];
  for (int i = t; i < NA; i += 256)     hMask[i] = maskg[b * NA + i];
  __syncthreads();

  float msum = 0.f;
  #pragma unroll
  for (int n = 0; n < NA; ++n) msum += hMask[n];
  const float scale = rsqrtf(msum);   // (sum mask)^-0.5

  for (int p = t; p < NA * NA; p += 256) {
    int a = p >> 5, n = p & 31;
    float dx = hGeo[n * 3 + 0] - hGeo[a * 3 + 0];
    float dy = hGeo[n * 3 + 1] - hGeo[a * 3 + 1];
    float dz = hGeo[n * 3 + 2] - hGeo[a * 3 + 2];
    float rad = sqrtf(dx * dx + dy * dy + dz * dz + 1e-12f);
    float inv = 1.f / (rad + 1e-8f);
    hRhat[(a * 3 + 0) * NA + n] = dx * inv;
    hRhat[(a * 3 + 1) * NA + n] = dy * inv;
    hRhat[(a * 3 + 2) * NA + n] = dz * inv;
    float mn = hMask[n];
    #pragma unroll
    for (int k = 0; k < NK; ++k) {
      float d = rad - (5.0f / 31.0f) * (float)k;   // linspace(0, RMAX, K)
      hG[(a * NA + n) * NK + k] = (_Float16)(__expf(-10.f * d * d) * mn);
    }
  }
  // s1 = emb[Z] * scale, stored transposed sT[c][n] (f32: contraction operand only)
  for (int e = t; e < NE * NA; e += 256) {
    int c = e >> 5, n = e & 31;
    int z = Z[b * NA + n];
    hSf[c * NA + n] = emb[z * NE + c] * scale;
  }
  // layer-1 radial weights, transposed: cols 0..63 = ss, 64..127 = sv
  for (int e = t; e < 128 * NK; e += 256) {
    int c = e >> 5, k = e & 31;
    float wv = (c < 64) ? Wr_ss1[k * 64 + c] : Wr_sv1[k * 64 + (c - 64)];
    hWcat[c * NK + k] = (_Float16)wv;
  }
  // layer-1 mix weights, transposed [co][ci]
  for (int e = t; e < 80 * 64; e += 256) { int co = e >> 6, ci = e & 63; hWmSS[co * 64 + ci] = (_Float16)Wmix_ss1[ci * 80 + co]; }
  for (int e = t; e < 16 * 64; e += 256) { int co = e >> 6, ci = e & 63; hWmSV[co * 64 + ci] = (_Float16)Wmix_sv1[ci * 16 + co]; }
  __syncthreads();

  // ======================= Layer 1: per-pair radial GEMMs =======================
  for (int a = w; a < NA; a += 8) {
    // A operands: G[a] rows n (M) x k (contraction 32), two 16-row tiles
    v16h A0 = ld_ab(&hG[(a * NA + m) * NK + kh],        &hG[(a * NA + m) * NK + kh + 16]);
    v16h A1 = ld_ab(&hG[(a * NA + 16 + m) * NK + kh],   &hG[(a * NA + 16 + m) * NK + kh + 16]);
    for (int ct = 0; ct < 8; ++ct) {
      int c = ct * 16 + m;
      v16h Bv = ld_ab(&hWcat[c * NK + kh], &hWcat[c * NK + kh + 16]);
      v8f D0 = wmma_f16(A0, Bv, z8);   // rows n=0..15
      v8f D1 = wmma_f16(A1, Bv, z8);   // rows n=16..31
      if (ct < 4) {          // ---- ss path: sum_n R[n,c]*s1[n,c]
        F8 s0  = ld_f8(&hSf[c * NA + 8 * h]);
        F8 s1v = ld_f8(&hSf[c * NA + 16 + 8 * h]);
        float pacc = 0.f;
        #pragma unroll
        for (int j = 0; j < 8; ++j)
          pacc += D0[j] * s0.f[j] + D1[j] * s1v.f[j];
        pacc += __shfl_xor(pacc, 16);
        if (h == 0) hPss[a * 64 + c] = (_Float16)pacc;
      } else {               // ---- sv path: sum_n R[n,c]*rhat[n,x]*s1[n,c]
        int cp = c - 64;
        F8 s0  = ld_f8(&hSf[cp * NA + 8 * h]);
        F8 s1v = ld_f8(&hSf[cp * NA + 16 + 8 * h]);
        #pragma unroll
        for (int x = 0; x < 3; ++x) {
          F8 r0 = ld_f8(&hRhat[(a * 3 + x) * NA + 8 * h]);
          F8 r1 = ld_f8(&hRhat[(a * 3 + x) * NA + 16 + 8 * h]);
          float pacc = 0.f;
          #pragma unroll
          for (int j = 0; j < 8; ++j)
            pacc += D0[j] * (s0.f[j] * r0.f[j]) + D1[j] * (s1v.f[j] * r1.f[j]);
          pacc += __shfl_xor(pacc, 16);
          if (h == 0) hPhv[(x * NA + a) * 64 + cp] = (_Float16)pacc;
        }
      }
    }
  }
  __syncthreads();

  // ======================= Layer 1: channel-mix GEMMs =======================
  for (int tt = w; tt < 16; tt += 8) {
    if (tt < 10) {           // s_out[32x80] = Pss[32x64] @ Wmix_ss1
      int mt = tt & 1, ct = tt >> 1;
      v8f D = z8;
      #pragma unroll
      for (int kp = 0; kp < 2; ++kp) {
        v16h A  = ld_ab(&hPss[(mt * 16 + m) * 64 + kp * 32 + kh],
                        &hPss[(mt * 16 + m) * 64 + kp * 32 + kh + 16]);
        v16h Bv = ld_ab(&hWmSS[(ct * 16 + m) * 64 + kp * 32 + kh],
                        &hWmSS[(ct * 16 + m) * 64 + kp * 32 + kh + 16]);
        D = wmma_f16(A, Bv, D);
      }
      #pragma unroll
      for (int j = 0; j < 8; ++j)
        hSoutF[(mt * 16 + j + 8 * h) * NSG + ct * 16 + m] = D[j];
    } else {                 // v_out_x[32x16] = Phv_x[32x64] @ Wmix_sv1
      int tv = tt - 10, x = tv >> 1, mt = tv & 1;
      v8f D = z8;
      #pragma unroll
      for (int kp = 0; kp < 2; ++kp) {
        v16h A  = ld_ab(&hPhv[(x * NA + mt * 16 + m) * 64 + kp * 32 + kh],
                        &hPhv[(x * NA + mt * 16 + m) * 64 + kp * 32 + kh + 16]);
        v16h Bv = ld_ab(&hWmSV[m * 64 + kp * 32 + kh],
                        &hWmSV[m * 64 + kp * 32 + kh + 16]);
        D = wmma_f16(A, Bv, D);
      }
      #pragma unroll
      for (int j = 0; j < 8; ++j)
        hVoutF[(x * NA + mt * 16 + j + 8 * h) * NL1 + m] = D[j];
    }
  }
  __syncthreads();

  // ======================= Gating + stage layer-2 inputs/weights =======================
  for (int e = t; e < NE * NA; e += 256) {      // s2 = silu(s_out)*mask*scale (transposed)
    int c = e >> 5, a = e & 31;
    hSf[c * NA + a] = siluf_(hSoutF[a * NSG + c]) * hMask[a] * scale;
  }
  for (int e = t; e < 3 * NL1 * NA; e += 256) { // v2 = v_out*sigmoid(gate)*mask*scale
    int x = e / (NL1 * NA); int rem = e % (NL1 * NA);
    int c16 = rem >> 5; int a = rem & 31;
    float g = sigmoidf_(hSoutF[a * NSG + 64 + c16]);
    hVf[(x * NL1 + c16) * NA + a] = hVoutF[(x * NA + a) * NL1 + c16] * g * hMask[a] * scale;
  }
  for (int e = t; e < 160 * NK; e += 256) {     // Wcat2: ss|sv|vv|vs
    int c = e >> 5, k = e & 31;
    float wv;
    if (c < 64)       wv = Wr_ss2[k * 64 + c];
    else if (c < 128) wv = Wr_sv2[k * 64 + (c - 64)];
    else if (c < 144) wv = Wr_vv2[k * 16 + (c - 128)];
    else              wv = Wr_vs2[k * 16 + (c - 144)];
    hWcat[c * NK + k] = (_Float16)wv;
  }
  for (int e = t; e < 80 * 64; e += 256) { int co = e >> 6, ci = e & 63; hWmSS[co * 64 + ci] = (_Float16)Wmix_ss2[ci * 80 + co]; }
  for (int e = t; e < 16 * 64; e += 256) { int co = e >> 6, ci = e & 63; hWmSV[co * 64 + ci] = (_Float16)Wmix_sv2[ci * 16 + co]; }
  for (int e = t; e < 16 * 32; e += 256) { int co = e >> 5, ci = e & 31; hWmVV[co * 32 + ci] = (_Float16)((ci < 16) ? Wmix_vv2[ci * 16 + co] : 0.f); }
  for (int e = t; e < 80 * 32; e += 256) { int co = e >> 5, ci = e & 31; hWmVS[co * 32 + ci] = (_Float16)((ci < 16) ? Wmix_vs2[ci * 80 + co] : 0.f); }
  // zero the k-padding of Pvv/Pvs (contraction padded 16 -> 32 for WMMA)
  for (int e = t; e < 3 * NA * 16; e += 256) { int row = e >> 4; int kk = e & 15; hPvv[row * 32 + 16 + kk] = (_Float16)0.f; }
  for (int e = t; e < NA * 16; e += 256)     { int row = e >> 4; int kk = e & 15; hPvs[row * 32 + 16 + kk] = (_Float16)0.f; }
  __syncthreads();

  // ======================= Layer 2: per-pair radial GEMMs =======================
  for (int a = w; a < NA; a += 8) {
    v16h A0 = ld_ab(&hG[(a * NA + m) * NK + kh],      &hG[(a * NA + m) * NK + kh + 16]);
    v16h A1 = ld_ab(&hG[(a * NA + 16 + m) * NK + kh], &hG[(a * NA + 16 + m) * NK + kh + 16]);
    for (int ct = 0; ct < 10; ++ct) {
      int c = ct * 16 + m;
      v16h Bv = ld_ab(&hWcat[c * NK + kh], &hWcat[c * NK + kh + 16]);
      v8f D0 = wmma_f16(A0, Bv, z8);
      v8f D1 = wmma_f16(A1, Bv, z8);
      if (ct < 4) {          // ss
        F8 s0  = ld_f8(&hSf[c * NA + 8 * h]);
        F8 s1v = ld_f8(&hSf[c * NA + 16 + 8 * h]);
        float pacc = 0.f;
        #pragma unroll
        for (int j = 0; j < 8; ++j)
          pacc += D0[j] * s0.f[j] + D1[j] * s1v.f[j];
        pacc += __shfl_xor(pacc, 16);
        if (h == 0) hPss[a * 64 + c] = (_Float16)pacc;
      } else if (ct < 8) {   // sv
        int cp = c - 64;
        F8 s0  = ld_f8(&hSf[cp * NA + 8 * h]);
        F8 s1v = ld_f8(&hSf[cp * NA + 16 + 8 * h]);
        #pragma unroll
        for (int x = 0; x < 3; ++x) {
          F8 r0 = ld_f8(&hRhat[(a * 3 + x) * NA + 8 * h]);
          F8 r1 = ld_f8(&hRhat[(a * 3 + x) * NA + 16 + 8 * h]);
          float pacc = 0.f;
          #pragma unroll
          for (int j = 0; j < 8; ++j)
            pacc += D0[j] * (s0.f[j] * r0.f[j]) + D1[j] * (s1v.f[j] * r1.f[j]);
          pacc += __shfl_xor(pacc, 16);
          if (h == 0) hPhv[(x * NA + a) * 64 + cp] = (_Float16)pacc;
        }
      } else if (ct == 8) {  // vv: sum_n R[n,c]*v2[n,c16,x]
        int c16 = m;
        #pragma unroll
        for (int x = 0; x < 3; ++x) {
          F8 v0 = ld_f8(&hVf[(x * NL1 + c16) * NA + 8 * h]);
          F8 v1 = ld_f8(&hVf[(x * NL1 + c16) * NA + 16 + 8 * h]);
          float pacc = 0.f;
          #pragma unroll
          for (int j = 0; j < 8; ++j)
            pacc += D0[j] * v0.f[j] + D1[j] * v1.f[j];
          pacc += __shfl_xor(pacc, 16);
          if (h == 0) hPvv[(x * NA + a) * 32 + c16] = (_Float16)pacc;
        }
      } else {               // vs: sum_n R[n,c]*(rhat[n,:]·v2[n,c16,:])
        int c16 = m;
        float pacc = 0.f;
        F8 rx0[3], rx1[3], vv0[3], vv1[3];
        #pragma unroll
        for (int x = 0; x < 3; ++x) {
          rx0[x] = ld_f8(&hRhat[(a * 3 + x) * NA + 8 * h]);
          rx1[x] = ld_f8(&hRhat[(a * 3 + x) * NA + 16 + 8 * h]);
          vv0[x] = ld_f8(&hVf[(x * NL1 + c16) * NA + 8 * h]);
          vv1[x] = ld_f8(&hVf[(x * NL1 + c16) * NA + 16 + 8 * h]);
        }
        #pragma unroll
        for (int j = 0; j < 8; ++j) {
          float rd0 = rx0[0].f[j] * vv0[0].f[j] + rx0[1].f[j] * vv0[1].f[j] + rx0[2].f[j] * vv0[2].f[j];
          float rd1 = rx1[0].f[j] * vv1[0].f[j] + rx1[1].f[j] * vv1[1].f[j] + rx1[2].f[j] * vv1[2].f[j];
          pacc += D0[j] * rd0 + D1[j] * rd1;
        }
        pacc += __shfl_xor(pacc, 16);
        if (h == 0) hPvs[a * 32 + c16] = (_Float16)pacc;
      }
    }
  }
  __syncthreads();

  // ======================= Layer 2: channel-mix GEMMs =======================
  for (int tt = w; tt < 16; tt += 8) {
    if (tt < 10) {           // s_out = Pss@Wmix_ss2 + Pvs@Wmix_vs2
      int mt = tt & 1, ct = tt >> 1;
      v8f D = z8;
      #pragma unroll
      for (int kp = 0; kp < 2; ++kp) {
        v16h A  = ld_ab(&hPss[(mt * 16 + m) * 64 + kp * 32 + kh],
                        &hPss[(mt * 16 + m) * 64 + kp * 32 + kh + 16]);
        v16h Bv = ld_ab(&hWmSS[(ct * 16 + m) * 64 + kp * 32 + kh],
                        &hWmSS[(ct * 16 + m) * 64 + kp * 32 + kh + 16]);
        D = wmma_f16(A, Bv, D);
      }
      {
        v16h A  = ld_ab(&hPvs[(mt * 16 + m) * 32 + kh], &hPvs[(mt * 16 + m) * 32 + kh + 16]);
        v16h Bv = ld_ab(&hWmVS[(ct * 16 + m) * 32 + kh], &hWmVS[(ct * 16 + m) * 32 + kh + 16]);
        D = wmma_f16(A, Bv, D);
      }
      #pragma unroll
      for (int j = 0; j < 8; ++j)
        hSoutF[(mt * 16 + j + 8 * h) * NSG + ct * 16 + m] = D[j];
    } else {                 // v_out_x = Phv_x@Wmix_sv2 + Pvv_x@Wmix_vv2
      int tv = tt - 10, x = tv >> 1, mt = tv & 1;
      v8f D = z8;
      #pragma unroll
      for (int kp = 0; kp < 2; ++kp) {
        v16h A  = ld_ab(&hPhv[(x * NA + mt * 16 + m) * 64 + kp * 32 + kh],
                        &hPhv[(x * NA + mt * 16 + m) * 64 + kp * 32 + kh + 16]);
        v16h Bv = ld_ab(&hWmSV[m * 64 + kp * 32 + kh],
                        &hWmSV[m * 64 + kp * 32 + kh + 16]);
        D = wmma_f16(A, Bv, D);
      }
      {
        v16h A  = ld_ab(&hPvv[(x * NA + mt * 16 + m) * 32 + kh],
                        &hPvv[(x * NA + mt * 16 + m) * 32 + kh + 16]);
        v16h Bv = ld_ab(&hWmVV[m * 32 + kh], &hWmVV[m * 32 + kh + 16]);
        D = wmma_f16(A, Bv, D);
      }
      #pragma unroll
      for (int j = 0; j < 8; ++j)
        hVoutF[(x * NA + mt * 16 + j + 8 * h) * NL1 + m] = D[j];
    }
  }
  __syncthreads();

  // ======================= Final gating + store =======================
  for (int e = t; e < NA * NOUT; e += 256) {
    int a = e / NOUT, c = e % NOUT;
    float val;
    if (c < NL0) {
      val = siluf_(hSoutF[a * NSG + c]) * hMask[a];
    } else {
      int cc = c - NL0, c16 = cc / 3, x = cc % 3;
      float g = sigmoidf_(hSoutF[a * NSG + 64 + c16]);
      val = hVoutF[(x * NA + a) * NL1 + c16] * g * hMask[a];
    }
    out[(b * NA + a) * NOUT + c] = val;
  }
}

extern "C" void kernel_launch(void* const* d_in, const int* in_sizes, int n_in,
                              void* d_out, int out_size, void* d_ws, size_t ws_size,
                              hipStream_t stream) {
  (void)in_sizes; (void)n_in; (void)out_size; (void)d_ws; (void)ws_size;
  const int*   Z        = (const int*)d_in[0];
  const float* geometry = (const float*)d_in[1];
  const float* mask     = (const float*)d_in[2];
  const float* emb      = (const float*)d_in[3];
  const float* Wr_ss1   = (const float*)d_in[4];
  const float* Wmix_ss1 = (const float*)d_in[5];
  const float* Wr_sv1   = (const float*)d_in[6];
  const float* Wmix_sv1 = (const float*)d_in[7];
  const float* Wr_ss2   = (const float*)d_in[8];
  const float* Wmix_ss2 = (const float*)d_in[9];
  const float* Wr_sv2   = (const float*)d_in[10];
  const float* Wmix_sv2 = (const float*)d_in[11];
  const float* Wr_vv2   = (const float*)d_in[12];
  const float* Wmix_vv2 = (const float*)d_in[13];
  const float* Wr_vs2   = (const float*)d_in[14];
  const float* Wmix_vs2 = (const float*)d_in[15];
  float* out = (float*)d_out;

  painn2_kernel<<<dim3(NB), dim3(256), SMEM_BYTES, stream>>>(
      Z, geometry, mask, emb,
      Wr_ss1, Wmix_ss1, Wr_sv1, Wmix_sv1,
      Wr_ss2, Wmix_ss2, Wr_sv2, Wmix_sv2,
      Wr_vv2, Wmix_vv2, Wr_vs2, Wmix_vs2, out);
}
